// SageBase_13202729468517
// MI455X (gfx1250) — matrix operand order repaired and checked by reading, not compiled, not run on hardware
//
#include <hip/hip_runtime.h>
#include <math.h>

typedef __bf16 bf16_t;
typedef __attribute__((ext_vector_type(16))) __bf16 v16bf;
typedef __attribute__((ext_vector_type(8)))  __bf16 v8bf;
typedef __attribute__((ext_vector_type(8)))  float  v8f;

#define HID 256

// ---------------------------------------------------------------- utilities
__global__ void zero_f32_kernel(float* p, size_t n) {
    size_t i = (size_t)blockIdx.x * blockDim.x + threadIdx.x;
    size_t stride = (size_t)gridDim.x * blockDim.x;
    for (; i < n; i += stride) p[i] = 0.0f;
}

__global__ void zero_bf16_kernel(bf16_t* p, size_t n) {
    size_t i = (size_t)blockIdx.x * blockDim.x + threadIdx.x;
    size_t stride = (size_t)gridDim.x * blockDim.x;
    for (; i < n; i += stride) p[i] = (bf16_t)0.0f;
}

// ---------------------------------------------------------------- normalize
// stats layout (float buffer, some slots int-aliased; all source values > 0):
// [0]=minx(bits) [1]=maxx(bits) [2]=miny(bits) [3]=maxy(bits) [4]=sumx [5]=sumy [6]=maxa(bits)
__global__ void init_stats_kernel(float* stats) {
    int* si = (int*)stats;
    si[0] = __float_as_int(3.4e38f);
    si[1] = 0;
    si[2] = __float_as_int(3.4e38f);
    si[3] = 0;
    stats[4] = 0.0f;
    stats[5] = 0.0f;
    si[6] = 0;
}

__global__ void reduce_stats_kernel(const float* __restrict__ x, int N, float* stats) {
    int i = blockIdx.x * blockDim.x + threadIdx.x;
    int stride = gridDim.x * blockDim.x;
    float mnx = 3.4e38f, mxx = 0.0f, mny = 3.4e38f, mxy = 0.0f, mxa = 0.0f;
    float sx = 0.0f, sy = 0.0f;
    for (; i < N; i += stride) {
        float a = x[i * 3 + 0], b = x[i * 3 + 1], c = x[i * 3 + 2];
        mnx = fminf(mnx, a); mxx = fmaxf(mxx, a);
        mny = fminf(mny, b); mxy = fmaxf(mxy, b);
        sx += a; sy += b;
        mxa = fmaxf(mxa, c);
    }
    int* si = (int*)stats;
    atomicMin(&si[0], __float_as_int(mnx));
    atomicMax(&si[1], __float_as_int(mxx));
    atomicMin(&si[2], __float_as_int(mny));
    atomicMax(&si[3], __float_as_int(mxy));
    atomicAdd(&stats[4], sx);
    atomicAdd(&stats[5], sy);
    atomicMax(&si[6], __float_as_int(mxa));
}

// params: [0]=cond [1]=mean0 [2]=mean1 [3]=max0 [4]=max1 [5]=maxa
__global__ void finalize_stats_kernel(const float* stats, float* params, int N) {
    const int* si = (const int*)stats;
    float mnx = __int_as_float(si[0]), mxx = __int_as_float(si[1]);
    float mny = __int_as_float(si[2]), mxy = __int_as_float(si[3]);
    float sx = stats[4], sy = stats[5];
    float mxa = __int_as_float(si[6]);
    bool cond = (mxy - mny) > (mxx - mnx);
    float mean0, mean1, max0, max1;
    float invN = 1.0f / (float)N;
    if (cond) { // rotated coords: (c0,c1) = (-y, x)
        mean0 = -sy * invN; mean1 = sx * invN;
        max0  = -mny;       max1  = mxx;
    } else {
        mean0 = sx * invN;  mean1 = sy * invN;
        max0  = mxx;        max1  = mxy;
    }
    params[0] = cond ? 1.0f : 0.0f;
    params[1] = mean0; params[2] = mean1;
    params[3] = max0;  params[4] = max1;
    params[5] = mxa;
}

__global__ void normalize_kernel(const float* __restrict__ x, const float* __restrict__ params,
                                 float* __restrict__ h0, int N) {
    int n = blockIdx.x * blockDim.x + threadIdx.x;
    if (n >= N) return;
    float c0 = x[n * 3 + 0], c1 = x[n * 3 + 1], a = x[n * 3 + 2];
    if (params[0] != 0.0f) { float t = c0; c0 = -c1; c1 = t; }
    h0[n * 3 + 0] = (c0 - params[1]) / params[3];
    h0[n * 3 + 1] = (c1 - params[2]) / params[4];
    h0[n * 3 + 2] = a / params[5];
}

// ---------------------------------------------------------------- graph prep
__global__ void degree_kernel(const int* __restrict__ dst, float* __restrict__ cnt, int E) {
    int e = blockIdx.x * blockDim.x + threadIdx.x;
    if (e < E) atomicAdd(&cnt[dst[e]], 1.0f);
}

__global__ void inv_kernel(const float* __restrict__ cnt, float* __restrict__ inv, int n) {
    int i = blockIdx.x * blockDim.x + threadIdx.x;
    if (i < n) inv[i] = 1.0f / fmaxf(cnt[i], 1.0f);
}

__global__ void scatter3_kernel(const float* __restrict__ h0, const int* __restrict__ src,
                                const int* __restrict__ dst, float* __restrict__ agg0, int E) {
    int e = blockIdx.x * blockDim.x + threadIdx.x;
    if (e >= E) return;
    int s = src[e], d = dst[e];
    atomicAdd(&agg0[d * 3 + 0], h0[s * 3 + 0]);
    atomicAdd(&agg0[d * 3 + 1], h0[s * 3 + 1]);
    atomicAdd(&agg0[d * 3 + 2], h0[s * 3 + 2]);
}

// wave-per-edge scatter-add of 256-dim bf16 features into f32 accumulator
__global__ void scatter256_kernel(const bf16_t* __restrict__ H, const int* __restrict__ src,
                                  const int* __restrict__ dst, float* __restrict__ agg, int E) {
    int e = blockIdx.x * 8 + (threadIdx.x >> 5);
    if (e >= E) return;
    int lane = threadIdx.x & 31;
    const bf16_t* hs = H + (size_t)src[e] * HID;
    float* ad = agg + (size_t)dst[e] * HID;
#pragma unroll
    for (int i = 0; i < 8; ++i) {
        int k = lane + i * 32;
        atomicAdd(&ad[k], (float)hs[k]);
    }
}

__global__ void scale_to_bf16_kernel(const float* __restrict__ agg, const float* __restrict__ inv,
                                     bf16_t* __restrict__ out, size_t total) {
    size_t i = (size_t)blockIdx.x * blockDim.x + threadIdx.x;
    if (i < total) out[i] = (bf16_t)(agg[i] * inv[i >> 8]);
}

// ---------------------------------------------------------------- layer 1 (K=3, trivial)
__global__ void layer1_kernel(const float* __restrict__ h0, const float* __restrict__ agg0,
                              const float* __restrict__ inv, const float* __restrict__ w1l,
                              const float* __restrict__ b1, const float* __restrict__ w1r,
                              bf16_t* __restrict__ out, int N) {
    int idx = blockIdx.x * blockDim.x + threadIdx.x;
    int n = idx >> 8;
    int j = idx & 255;
    if (n >= N) return;
    float iv = inv[n];
    float acc = b1[j];
#pragma unroll
    for (int t = 0; t < 3; ++t) {
        acc += agg0[n * 3 + t] * iv * w1l[t * HID + j] + h0[n * 3 + t] * w1r[t * HID + j];
    }
    out[(size_t)n * HID + j] = (bf16_t)tanhf(acc);
}

// ---------------------------------------------------------------- weight packing
// Pack W (256x256 f32, row-major K x Cout) into bf16 B-fragment order:
// packed[((cb*8 + kb)*32 + lane)*16 + e]  where
//   col = cb*16 + (lane & 15),  k = kb*32 + (lane>>4)*8 + (e&7) + (e>>3)*16
__global__ void pack_weight_kernel(const float* __restrict__ w, bf16_t* __restrict__ bp) {
    int idx = blockIdx.x * blockDim.x + threadIdx.x; // 65536 total
    if (idx >= 256 * 256) return;
    int e    = idx & 15;
    int lane = (idx >> 4) & 31;
    int kb   = (idx >> 9) & 7;
    int cb   = idx >> 12;
    int c = cb * 16 + (lane & 15);
    int k = kb * 32 + (lane >> 4) * 8 + (e & 7) + (e >> 3) * 16;
    bp[idx] = (bf16_t)w[k * HID + c];
}

// ---------------------------------------------------------------- WMMA GEMM
// acc += A[rows, K=256] (row-major bf16) x packed-B   (4 col tiles per wave)
__device__ __forceinline__ void gemm_accum(const bf16_t* __restrict__ A,
                                           const bf16_t* __restrict__ Bp,
                                           size_t rowBase, int cbBase,
                                           int lane, int half, int lsub, v8f* acc) {
    const bf16_t* arow = A + (rowBase + (size_t)lsub) * HID;
#pragma unroll
    for (int kb = 0; kb < 8; ++kb) {
        const bf16_t* ap = arow + kb * 32 + half * 8;
        v8bf lo = *(const v8bf*)ap;         // K[0..7]  (lanes 0-15) / K[8..15]  (lanes 16-31)
        v8bf hi = *(const v8bf*)(ap + 16);  // K[16..23]            / K[24..31]
        union { v16bf v; v8bf h[2]; } ua;
        ua.h[0] = lo; ua.h[1] = hi;
        v16bf a = ua.v;
#pragma unroll
        for (int c = 0; c < 4; ++c) {
            const v16bf* bptr = (const v16bf*)Bp + (((size_t)(cbBase + c) * 8 + kb) * 32 + lane);
            v16bf b = *bptr;
            acc[c] = __builtin_amdgcn_wmma_f32_16x16x32_bf16(
                false, a, false, b, (short)0, acc[c], false, false);
        }
    }
}

// out = tanh(A1 @ W1 + [A2 @ W2] + bias); block = 8 waves, 128 rows x 64 cols
__global__ void gemm_tanh_kernel(const bf16_t* __restrict__ A1, const bf16_t* __restrict__ Bp1,
                                 const bf16_t* __restrict__ A2, const bf16_t* __restrict__ Bp2,
                                 const float* __restrict__ bias, bf16_t* __restrict__ out) {
    const int lane = threadIdx.x & 31;
    const int wave = threadIdx.x >> 5;
    const int half = lane >> 4;
    const int lsub = lane & 15;
    const size_t rowBase = (size_t)blockIdx.x * 128 + (size_t)wave * 16;
    const int cbBase = blockIdx.y * 4;

    v8f acc[4];
#pragma unroll
    for (int c = 0; c < 4; ++c)
#pragma unroll
        for (int i = 0; i < 8; ++i) acc[c][i] = 0.0f;

    gemm_accum(A1, Bp1, rowBase, cbBase, lane, half, lsub, acc);
    if (A2 != nullptr) gemm_accum(A2, Bp2, rowBase, cbBase, lane, half, lsub, acc);

#pragma unroll
    for (int c = 0; c < 4; ++c) {
        int col = (cbBase + c) * 16 + lsub;
        float bv = bias[col];
#pragma unroll
        for (int i = 0; i < 8; ++i) {
            float v = acc[c][i] + bv;
            out[(rowBase + (size_t)(i + 8 * half)) * HID + col] = (bf16_t)tanhf(v);
        }
    }
}

// ---------------------------------------------------------------- softmax head
__global__ void head_kernel(const bf16_t* __restrict__ H, const float* __restrict__ wlast,
                            const float* __restrict__ blast, float* __restrict__ out, int N) {
    int n = blockIdx.x * blockDim.x + threadIdx.x;
    if (n >= N) return;
    const bf16_t* h = H + (size_t)n * HID;
    float a0 = blast[0], a1 = blast[1], a2 = blast[2], a3 = blast[3];
    for (int k = 0; k < HID; ++k) {
        float hv = (float)h[k];
        a0 += hv * wlast[k * 4 + 0];
        a1 += hv * wlast[k * 4 + 1];
        a2 += hv * wlast[k * 4 + 2];
        a3 += hv * wlast[k * 4 + 3];
    }
    float m = fmaxf(fmaxf(a0, a1), fmaxf(a2, a3));
    float e0 = expf(a0 - m), e1 = expf(a1 - m), e2 = expf(a2 - m), e3 = expf(a3 - m);
    float s = 1.0f / (e0 + e1 + e2 + e3);
    out[n * 4 + 0] = e0 * s;
    out[n * 4 + 1] = e1 * s;
    out[n * 4 + 2] = e2 * s;
    out[n * 4 + 3] = e3 * s;
}

// ---------------------------------------------------------------- launch
extern "C" void kernel_launch(void* const* d_in, const int* in_sizes, int n_in,
                              void* d_out, int out_size, void* d_ws, size_t ws_size,
                              hipStream_t stream) {
    const float* x    = (const float*)d_in[0];
    const int*   edge = (const int*)d_in[1];
    const int N = in_sizes[0] / 3;
    const int E = in_sizes[1] / 2;
    const int* src = edge;
    const int* dst = edge + E;

    const float* w1l = (const float*)d_in[2];
    const float* b1  = (const float*)d_in[3];
    const float* w1r = (const float*)d_in[4];
    const float* wBiases[3] = { (const float*)d_in[6], (const float*)d_in[9], (const float*)d_in[12] };
    const float* wL[3] = { (const float*)d_in[5], (const float*)d_in[8],  (const float*)d_in[11] };
    const float* wR[3] = { (const float*)d_in[7], (const float*)d_in[10], (const float*)d_in[13] };
    const float* wl1 = (const float*)d_in[14];
    const float* bl1 = (const float*)d_in[15];
    const float* wl2 = (const float*)d_in[16];
    const float* bl2 = (const float*)d_in[17];
    const float* wlast = (const float*)d_in[18];
    const float* blast = (const float*)d_in[19];
    float* out = (float*)d_out;

    const size_t Npad = (size_t)((N + 127) / 128) * 128;
    const size_t featElems = Npad * HID;

    // workspace carve-up (256B aligned)
    size_t off = 0;
    auto carve = [&](size_t bytes) {
        void* p = (char*)d_ws + off;
        off += (bytes + 255) & ~(size_t)255;
        return p;
    };
    bf16_t* HA    = (bf16_t*)carve(featElems * sizeof(bf16_t));
    bf16_t* HB    = (bf16_t*)carve(featElems * sizeof(bf16_t));
    bf16_t* AGB   = (bf16_t*)carve(featElems * sizeof(bf16_t));
    float*  AGF   = (float*)carve(featElems * sizeof(float));
    float*  h0    = (float*)carve((size_t)N * 3 * sizeof(float));
    float*  agg0  = (float*)carve((size_t)N * 3 * sizeof(float));
    float*  cnt   = (float*)carve(Npad * sizeof(float));
    float*  invd  = (float*)carve(Npad * sizeof(float));
    float*  stats = (float*)carve(64 * sizeof(float));
    float*  params= (float*)carve(16 * sizeof(float));
    bf16_t* BP[8];
    for (int i = 0; i < 8; ++i) BP[i] = (bf16_t*)carve((size_t)HID * HID * sizeof(bf16_t));
    (void)ws_size; (void)n_in; (void)out_size;

    const int T = 256;
    dim3 blk(T);

    // ---- normalization
    init_stats_kernel<<<1, 1, 0, stream>>>(stats);
    reduce_stats_kernel<<<256, blk, 0, stream>>>(x, N, stats);
    finalize_stats_kernel<<<1, 1, 0, stream>>>(stats, params, N);
    normalize_kernel<<<(N + T - 1) / T, blk, 0, stream>>>(x, params, h0, N);

    // ---- degrees
    zero_f32_kernel<<<1024, blk, 0, stream>>>(cnt, Npad);
    degree_kernel<<<(E + T - 1) / T, blk, 0, stream>>>(dst, cnt, E);
    inv_kernel<<<((int)Npad + T - 1) / T, blk, 0, stream>>>(cnt, invd, (int)Npad);

    // ---- layer 1 (K=3)
    zero_f32_kernel<<<1024, blk, 0, stream>>>(agg0, (size_t)N * 3);
    scatter3_kernel<<<(E + T - 1) / T, blk, 0, stream>>>(h0, src, dst, agg0, E);
    zero_bf16_kernel<<<2048, blk, 0, stream>>>(HA, featElems);
    zero_bf16_kernel<<<2048, blk, 0, stream>>>(HB, featElems);
    {
        long long tot = (long long)N * HID;
        layer1_kernel<<<(unsigned)((tot + T - 1) / T), blk, 0, stream>>>(
            h0, agg0, invd, w1l, b1, w1r, HA, N);
    }

    // ---- pack all 256x256 weights to WMMA fragment order
    const float* packSrc[8] = { wL[0], wR[0], wL[1], wR[1], wL[2], wR[2], wl1, wl2 };
    for (int i = 0; i < 8; ++i)
        pack_weight_kernel<<<(256 * 256) / T, blk, 0, stream>>>(packSrc[i], BP[i]);

    // ---- SAGE layers 2..4
    bf16_t* cur = HA;
    bf16_t* nxt = HB;
    dim3 ggrid((unsigned)(Npad / 128), 4);
    for (int L = 0; L < 3; ++L) {
        zero_f32_kernel<<<4096, blk, 0, stream>>>(AGF, featElems);
        scatter256_kernel<<<(E + 7) / 8, blk, 0, stream>>>(cur, src, dst, AGF, E);
        scale_to_bf16_kernel<<<(unsigned)((featElems + T - 1) / T), blk, 0, stream>>>(
            AGF, invd, AGB, featElems);
        gemm_tanh_kernel<<<ggrid, blk, 0, stream>>>(
            AGB, BP[2 * L], cur, BP[2 * L + 1], wBiases[L], nxt);
        bf16_t* t = cur; cur = nxt; nxt = t;
    }

    // ---- dense layers
    gemm_tanh_kernel<<<ggrid, blk, 0, stream>>>(cur, BP[6], nullptr, nullptr, bl1, nxt);
    { bf16_t* t = cur; cur = nxt; nxt = t; }
    gemm_tanh_kernel<<<ggrid, blk, 0, stream>>>(cur, BP[7], nullptr, nullptr, bl2, nxt);
    { bf16_t* t = cur; cur = nxt; nxt = t; }

    // ---- head + softmax
    head_kernel<<<(N + T - 1) / T, blk, 0, stream>>>(cur, wlast, blast, out, N);
}
